// GAT_simple_18107582120396
// MI455X (gfx1250) — matrix operand order repaired and checked by reading, not compiled, not run on hardware
//
#include <hip/hip_runtime.h>
#include <hip/hip_bf16.h>

typedef __attribute__((ext_vector_type(16))) _Float16 v16h;
typedef __attribute__((ext_vector_type(8)))  float    v8f;

#define LRELU_SLOPE 0.2f

__device__ __forceinline__ unsigned gat_f2key(float f) {
  unsigned u = __float_as_uint(f);
  return (u & 0x80000000u) ? ~u : (u | 0x80000000u);
}
__device__ __forceinline__ float gat_key2f(unsigned k) {
  unsigned u = (k & 0x80000000u) ? (k & 0x7FFFFFFFu) : ~k;
  return __uint_as_float(u);
}

__global__ void gat_zero_u32(unsigned* __restrict__ p, long n) {
  long i = (long)blockIdx.x * blockDim.x + threadIdx.x;
  if (i < n) p[i] = 0u;
}

// Pack W[K,N] (f32 row-major) into WMMA B-fragment order (f16):
// Bp[((nt*KSTEPS + ks)*32 + lane)*16 + 2v + {0,1}]
//   = W[(ks*32 + hi*8 + 2v + (v>=4 ? 8 : 0) + {0,1}) * N + nt*16 + (lane&15)]
// so the GEMM reads each per-lane fragment as one contiguous 32B chunk.
__global__ void gat_pack_b(const float* __restrict__ W, _Float16* __restrict__ Bp,
                           int K, int N) {
  int t = blockIdx.x * blockDim.x + threadIdx.x;
  int ksteps = K >> 5;
  int total = (N >> 4) * ksteps * 32;
  if (t >= total) return;
  int lane = t & 31;
  int ks = (t >> 5) % ksteps;
  int nt = (t >> 5) / ksteps;
  int hi = lane >> 4;
  int col = nt * 16 + (lane & 15);
  _Float16* o = Bp + (size_t)t * 16;
#pragma unroll
  for (int v = 0; v < 8; ++v) {
    int kb = ks * 32 + (hi ? 8 : 0) + 2 * v + (v >= 4 ? 8 : 0);
    o[2 * v]     = (_Float16)W[(size_t)kb * N + col];
    o[2 * v + 1] = (_Float16)W[(size_t)(kb + 1) * N + col];
  }
}

__device__ __forceinline__ void gat_cvt4(v16h& a, int base, float4 f) {
  a[base + 0] = (_Float16)f.x; a[base + 1] = (_Float16)f.y;
  a[base + 2] = (_Float16)f.z; a[base + 3] = (_Float16)f.w;
}

// C[M,N](f32) = A[M,K](f32, cvt->f16 in-register) * B(packed f16 fragments).
// One wave32 computes a 16 x (16*NT) strip: A fragment reused NT times.
// Requires M%16==0. K,N compile-time; k-loop fully unrolled.
template <int K, int N, int NT>
__global__ __launch_bounds__(256) void gat_gemm_wmma(
    const float* __restrict__ A, const _Float16* __restrict__ Bp,
    float* __restrict__ C, int M, int totalWarps) {
  constexpr int KSTEPS = K / 32;
  constexpr int NGROUPS = (N / 16) / NT;
  int warpId = (int)((blockIdx.x * blockDim.x + threadIdx.x) >> 5);
  if (warpId >= totalWarps) return;      // wave-uniform: EXEC all-1s for WMMA
  int lane = threadIdx.x & 31;
  int mt = warpId / NGROUPS;
  int gg = warpId - mt * NGROUPS;
  int l15 = lane & 15;
  int hi  = lane >> 4;                   // 0: lanes 0-15, 1: lanes 16-31
  const float* Ap = A + (size_t)(mt * 16 + l15) * K;
  v8f acc[NT];
#pragma unroll
  for (int t = 0; t < NT; ++t) acc[t] = (v8f){};
#pragma unroll
  for (int ks = 0; ks < KSTEPS; ++ks) {
    // A fragment (ISA 7.12.2 16-bit A 16x32): lanes 16-31 shifted +8/+24.
    const float4* a0 = reinterpret_cast<const float4*>(Ap + ks * 32 + hi * 8);
    const float4* a1 = reinterpret_cast<const float4*>(Ap + ks * 32 + 16 + hi * 8);
    float4 f0 = a0[0], f1 = a0[1], f2 = a1[0], f3 = a1[1];
    v16h a;
    gat_cvt4(a, 0, f0); gat_cvt4(a, 4, f1);
    gat_cvt4(a, 8, f2); gat_cvt4(a, 12, f3);
#pragma unroll
    for (int t = 0; t < NT; ++t) {
      int nt = gg * NT + t;
      v16h b = *reinterpret_cast<const v16h*>(
          Bp + ((size_t)(nt * KSTEPS + ks) * 32 + lane) * 16);
      acc[t] = __builtin_amdgcn_wmma_f32_16x16x32_f16(false, a, false, b,
                                                      (short)0, acc[t], false, false);
    }
  }
  // C/D layout: VGPR r -> M = r + 8*hi, N = lane&15
#pragma unroll
  for (int t = 0; t < NT; ++t) {
    int nt = gg * NT + t;
#pragma unroll
    for (int r = 0; r < 8; ++r) {
      int row = mt * 16 + r + (hi ? 8 : 0);
      C[(size_t)row * N + nt * 16 + l15] = acc[t][r];
    }
  }
}

// e_src[n,h] = <h[n,h,:], a_s[h,:]>, e_dst likewise. h layout [N,H,32].
__global__ void gat_scores(const float* __restrict__ h, const float* __restrict__ a_s,
                           const float* __restrict__ a_d, float* __restrict__ es,
                           float* __restrict__ ed, int NH, int H) {
  int t = blockIdx.x * blockDim.x + threadIdx.x;
  if (t >= NH) return;
  int hh = t % H;
  const float* hp = h + (long)t * 32;
  const float* sp = a_s + hh * 32;
  const float* dp = a_d + hh * 32;
  float s1 = 0.f, s2 = 0.f;
#pragma unroll
  for (int c = 0; c < 32; ++c) { float v = hp[c]; s1 += v * sp[c]; s2 += v * dp[c]; }
  es[t] = s1; ed[t] = s2;
}

// Edge set = E real edges + N self-loops (PyG GATConv default).
__global__ void gat_edge_max(const int* __restrict__ ei, int E_, int Nn,
                             const float* __restrict__ es, const float* __restrict__ ed,
                             unsigned* __restrict__ mkey, int H) {
  long t = (long)blockIdx.x * blockDim.x + threadIdx.x;
  long Et = (long)E_ + Nn;
  if (t >= Et * H) return;
  int e = (int)(t / H), hh = (int)(t - (long)e * H);
  int s = (e < E_) ? ei[e]      : (e - E_);
  int d = (e < E_) ? ei[E_ + e] : (e - E_);
  float a = es[s * H + hh] + ed[d * H + hh];
  a = a > 0.f ? a : LRELU_SLOPE * a;
  atomicMax(&mkey[d * H + hh], gat_f2key(a));
}

__global__ void gat_edge_sum(const int* __restrict__ ei, int E_, int Nn,
                             const float* __restrict__ es, const float* __restrict__ ed,
                             const unsigned* __restrict__ mkey, float* __restrict__ den, int H) {
  long t = (long)blockIdx.x * blockDim.x + threadIdx.x;
  long Et = (long)E_ + Nn;
  if (t >= Et * H) return;
  int e = (int)(t / H), hh = (int)(t - (long)e * H);
  int s = (e < E_) ? ei[e]      : (e - E_);
  int d = (e < E_) ? ei[E_ + e] : (e - E_);
  float a = es[s * H + hh] + ed[d * H + hh];
  a = a > 0.f ? a : LRELU_SLOPE * a;
  float m = gat_key2f(mkey[d * H + hh]);
  atomicAdd(&den[d * H + hh], __expf(a - m));
}

// One wave32 per edge: lane = channel (C==32), loop over heads.
__global__ void gat_edge_agg(const int* __restrict__ ei, int E_, int Nn,
                             const float* __restrict__ h, const float* __restrict__ es,
                             const float* __restrict__ ed, const unsigned* __restrict__ mkey,
                             const float* __restrict__ den, float* __restrict__ agg, int H) {
  int gw = (int)((blockIdx.x * blockDim.x + threadIdx.x) >> 5);
  int Et = E_ + Nn;
  if (gw >= Et) return;
  int lane = threadIdx.x & 31;
  int s = (gw < E_) ? ei[gw]      : (gw - E_);
  int d = (gw < E_) ? ei[E_ + gw] : (gw - E_);
  for (int hh = 0; hh < H; ++hh) {
    float a = es[s * H + hh] + ed[d * H + hh];
    a = a > 0.f ? a : LRELU_SLOPE * a;
    float m = gat_key2f(mkey[d * H + hh]);
    float coef = __expf(a - m) / (den[d * H + hh] + 1e-16f);
    atomicAdd(&agg[((long)d * H + hh) * 32 + lane],
              coef * h[((long)s * H + hh) * 32 + lane]);
  }
}

__global__ void gat_bias_elu(float* __restrict__ y, const float* __restrict__ bias,
                             long n, int D) {
  long t = (long)blockIdx.x * blockDim.x + threadIdx.x;
  if (t >= n) return;
  float v = y[t] + bias[(int)(t % D)];
  y[t] = v > 0.f ? v : expm1f(v);
}

__global__ void gat_pool(const float* __restrict__ h, const int* __restrict__ batch,
                         float* __restrict__ pooled, float* __restrict__ cnt, int Nn) {
  long t = (long)blockIdx.x * blockDim.x + threadIdx.x;
  if (t >= (long)Nn * 32) return;
  int n = (int)(t >> 5), c = (int)(t & 31);
  int g = batch[n];
  atomicAdd(&pooled[g * 32 + c], h[t]);
  if (c == 0) atomicAdd(&cnt[g], 1.0f);
}

__global__ void gat_pool_div(float* __restrict__ pooled, const float* __restrict__ cnt, int Gn) {
  int t = blockIdx.x * blockDim.x + threadIdx.x;
  if (t >= Gn * 32) return;
  pooled[t] /= fmaxf(cnt[t >> 5], 1.0f);
}

// Y[g,j] = bn(relu(X[g,:] @ W[:,j] + bias[j]))
__global__ void gat_fc_bn(const float* __restrict__ X, const float* __restrict__ W,
                          const float* __restrict__ bias, const float* __restrict__ gg,
                          const float* __restrict__ bb, const float* __restrict__ mm,
                          const float* __restrict__ vv, float* __restrict__ Y,
                          int Fin, int Fout) {
  int row = blockIdx.x, j = threadIdx.x;
  if (j >= Fout) return;
  float acc = bias[j];
  for (int k = 0; k < Fin; ++k) acc += X[row * Fin + k] * W[k * Fout + j];
  acc = fmaxf(acc, 0.f);
  Y[row * Fout + j] = (acc - mm[j]) * rsqrtf(vv[j] + 1e-5f) * gg[j] + bb[j];
}

__global__ void gat_head_out(const float* __restrict__ X, const float* __restrict__ w,
                             const float* __restrict__ b, float* __restrict__ out) {
  __shared__ float sbuf[128];
  int g = blockIdx.x, j = threadIdx.x;
  sbuf[j] = X[g * 128 + j] * w[j];
  __syncthreads();
  for (int st = 64; st > 0; st >>= 1) {
    if (j < st) sbuf[j] += sbuf[j + st];
    __syncthreads();
  }
  if (j == 0) out[g] = sbuf[0] + b[0];
}

extern "C" void kernel_launch(void* const* d_in, const int* in_sizes, int n_in,
                              void* d_out, int out_size, void* d_ws, size_t ws_size,
                              hipStream_t stream) {
  const float* x      = (const float*)d_in[0];
  const int*   ei     = (const int*)d_in[1];
  const int*   batch  = (const int*)d_in[2];
  const float* w_in   = (const float*)d_in[3];
  const float* as_in  = (const float*)d_in[4];
  const float* ad_in  = (const float*)d_in[5];
  const float* b_in   = (const float*)d_in[6];
  const float* w_mid  = (const float*)d_in[7];
  const float* as_mid = (const float*)d_in[8];
  const float* ad_mid = (const float*)d_in[9];
  const float* b_mid  = (const float*)d_in[10];
  const float* w_og   = (const float*)d_in[11];
  const float* as_og  = (const float*)d_in[12];
  const float* ad_og  = (const float*)d_in[13];
  const float* b_og   = (const float*)d_in[14];
  const float* fc1_w  = (const float*)d_in[15];
  const float* fc1_b  = (const float*)d_in[16];
  const float* fc2_w  = (const float*)d_in[17];
  const float* fc2_b  = (const float*)d_in[18];
  const float* bn1_g  = (const float*)d_in[19];
  const float* bn1_b  = (const float*)d_in[20];
  const float* bn1_m  = (const float*)d_in[21];
  const float* bn1_v  = (const float*)d_in[22];
  const float* bn2_g  = (const float*)d_in[23];
  const float* bn2_b  = (const float*)d_in[24];
  const float* bn2_m  = (const float*)d_in[25];
  const float* bn2_v  = (const float*)d_in[26];
  const float* out_w  = (const float*)d_in[27];
  const float* out_b  = (const float*)d_in[28];
  float* out = (float*)d_out;

  const int Nn = in_sizes[2];       // 50000 (divisible by 16)
  const int Ee = in_sizes[1] / 2;   // 800000
  const int Gn = out_size;          // 64
  (void)n_in; (void)ws_size;

  // ---- workspace carving (~109 MB) ----
  char* wsb = (char*)d_ws;
  size_t off = 0;
  auto carve = [&](size_t bytes) -> void* {
    void* p = wsb + off;
    off = (off + bytes + 255) & ~(size_t)255;
    return p;
  };
  float*    h0     = (float*)carve((size_t)Nn * 256 * 4);   // GEMM output h [N,H,32]
  float*    agg    = (float*)carve((size_t)Nn * 256 * 4);   // aggregated / next input
  _Float16* w16    = (_Float16*)carve((size_t)256 * 256 * 2); // packed B fragments
  float*    es     = (float*)carve((size_t)Nn * 8 * 4);
  float*    ed     = (float*)carve((size_t)Nn * 8 * 4);
  unsigned* mkey   = (unsigned*)carve((size_t)Nn * 8 * 4);
  float*    den    = (float*)carve((size_t)Nn * 8 * 4);
  float*    pooled = (float*)carve((size_t)Gn * 32 * 4);
  float*    cnt    = (float*)carve((size_t)Gn * 4);
  float*    t1     = (float*)carve((size_t)Gn * 128 * 4);
  float*    t2     = (float*)carve((size_t)Gn * 128 * 4);

  const int TB = 256;
  auto nb = [](long n, int tb) { return (unsigned)((n + tb - 1) / tb); };

  auto pack = [&](const float* W, int K, int N) {
    int total = (N >> 4) * (K >> 5) * 32;
    gat_pack_b<<<nb(total, TB), TB, 0, stream>>>(W, w16, K, N);
  };

  // Attention + softmax + scatter + bias/ELU (shared across layers).
  auto attention = [&](const float* a_s, const float* a_d, const float* bias, int H) {
    const int D = H * 32;
    gat_scores<<<nb((long)Nn * H, TB), TB, 0, stream>>>(h0, a_s, a_d, es, ed, Nn * H, H);
    gat_zero_u32<<<nb((long)Nn * D, TB), TB, 0, stream>>>((unsigned*)agg, (long)Nn * D);
    gat_zero_u32<<<nb((long)Nn * H, TB), TB, 0, stream>>>(mkey, (long)Nn * H);
    gat_zero_u32<<<nb((long)Nn * H, TB), TB, 0, stream>>>((unsigned*)den, (long)Nn * H);
    const long Et = (long)Ee + Nn;
    gat_edge_max<<<nb(Et * H, TB), TB, 0, stream>>>(ei, Ee, Nn, es, ed, mkey, H);
    gat_edge_sum<<<nb(Et * H, TB), TB, 0, stream>>>(ei, Ee, Nn, es, ed, mkey, den, H);
    gat_edge_agg<<<nb(Et * 32, TB), TB, 0, stream>>>(ei, Ee, Nn, h0, es, ed, mkey, den, agg, H);
    gat_bias_elu<<<nb((long)Nn * D, TB), TB, 0, stream>>>(agg, bias, (long)Nn * D, D);
  };

  // ---- input layer: [N,128] -> [N,256], 8 heads ----
  pack(w_in, 128, 256);
  {
    int tw = (Nn / 16) * ((256 / 16) / 4);
    gat_gemm_wmma<128, 256, 4><<<nb((long)tw * 32, TB), TB, 0, stream>>>(x, w16, h0, Nn, tw);
  }
  attention(as_in, ad_in, b_in, 8);

  // ---- middle layers: [N,256] -> [N,256] ----
  for (int i = 0; i < 2; ++i) {
    pack(w_mid + (size_t)i * 256 * 256, 256, 256);
    int tw = (Nn / 16) * ((256 / 16) / 4);
    // GEMM consumes agg (prev output) before attention() zeroes it (stream order).
    gat_gemm_wmma<256, 256, 4><<<nb((long)tw * 32, TB), TB, 0, stream>>>(agg, w16, h0, Nn, tw);
    attention(as_mid + i * 256, ad_mid + i * 256, b_mid + i * 256, 8);
  }

  // ---- output layer: [N,256] -> [N,32], 1 head, concat=False (mean of 1 head == id) ----
  pack(w_og, 256, 32);
  {
    int tw = (Nn / 16) * ((32 / 16) / 2);
    gat_gemm_wmma<256, 32, 2><<<nb((long)tw * 32, TB), TB, 0, stream>>>(agg, w16, h0, Nn, tw);
  }
  attention(as_og, ad_og, b_og, 1);

  // ---- global mean pool over graphs ----
  gat_zero_u32<<<nb((long)Gn * 32, TB), TB, 0, stream>>>((unsigned*)pooled, (long)Gn * 32);
  gat_zero_u32<<<nb((long)Gn, TB), TB, 0, stream>>>((unsigned*)cnt, (long)Gn);
  gat_pool<<<nb((long)Nn * 32, TB), TB, 0, stream>>>(agg, batch, pooled, cnt, Nn);
  gat_pool_div<<<nb((long)Gn * 32, TB), TB, 0, stream>>>(pooled, cnt, Gn);

  // ---- MLP head ----
  gat_fc_bn<<<Gn, 128, 0, stream>>>(pooled, fc1_w, fc1_b, bn1_g, bn1_b, bn1_m, bn1_v, t1, 32, 128);
  gat_fc_bn<<<Gn, 128, 0, stream>>>(t1, fc2_w, fc2_b, bn2_g, bn2_b, bn2_m, bn2_v, t2, 128, 128);
  gat_head_out<<<Gn, 128, 0, stream>>>(t2, out_w, out_b, out);
}
